// ASTGCNBlock_68504728371824
// MI455X (gfx1250) — matrix operand order repaired and checked by reading, not compiled, not run on hardware
//
#include <hip/hip_runtime.h>
#include <cmath>

// ASTGCN block for MI455X (gfx1250): one workgroup per batch element,
// x[b] staged in LDS, all big contractions via v_wmma_f32_16x16x32_f16.
// All WMMA operand buffers are stored PRE-SWIZZLED in the ISA per-lane
// fragment layout, so a fragment load is one 32B contiguous LDS read/lane.

#define B_   4096
#define N_   38
#define F_   64
#define T_   5
#define KCH  3
#define C_   64
#define NP   48      // padded node dim (3x16)
#define RP   192     // padded (n,t) rows (12x16)
#define FT   320     // F_*T_ columns (20x16)
#define KF   192     // KCH*F_ reduction dim (6x32)
#define EPSL 1e-5f

typedef _Float16 half_t;
typedef __attribute__((ext_vector_type(16))) _Float16 v16h;
typedef __attribute__((ext_vector_type(8)))  float    v8f;

// ---- LDS arena (bytes, all 32B aligned) ----
#define OFF_PF   0        // float[12288]: x[b] (38x320), later h (192x64)   49152
#define OFF_SAT  49152    // float[1444]: spatial attention S -> s_at         5792
#define OFF_SP   54944    // float[1444]: sp scratch                          5792
#define OFF_MISC 60736    // float[1408]: small attention scratch             5632
#define OFF_XH   66368    // half[64*320] swizzled B: f16 x                  40960
#define OFF_YB   107328   // half[5*48*192] swizzled A: Y_k per t            92160
#define OFF_AG   199488   // half[192*192] swizzled A: temporal-conv im2col  73728
#define OFF_TH   273216   // half[192*64] swizzled B: Theta, then W_time     24576
#define OFF_WRB  297792   // half[64*64] swizzled B: W_res                    8192
#define OFF_MH   305984   // half[48*64] swizzled A: (cheb_k*s_at)^T          6144
#define SMEM_SZ  312128

// ---- fragment-layout swizzles (CDNA5 WMMA 16x16x32 f16, wave32) ----
// A element (m,k): lane = ((k>>3)&1)*16 + (m&15), slot = ((k>>4)&1)*8 + (k&7)
__device__ inline int aswz(int m, int k, int ktiles) {
    int lane = ((k >> 3) & 1) * 16 + (m & 15);
    int pos  = ((k >> 4) & 1) * 8 + (k & 7);
    return (((((m >> 4) * ktiles) + (k >> 5)) * 32 + lane) << 4) + pos;
}
// B element (k,n): lane = ((k>>3)&1)*16 + (n&15), slot = ((k>>4)&1)*8 + (k&7)
__device__ inline int bswz(int k, int n, int ntiles) {
    int lane = ((k >> 3) & 1) * 16 + (n & 15);
    return (((((k >> 5) * ntiles) + (n >> 4)) * 32 + lane) << 4) + ((k >> 4) & 1) * 8 + (k & 7);
}
// fragment load: 32 contiguous bytes per lane (2x ds_load_b128)
__device__ inline v16h load_frag(const half_t* base, int tile32) {
    return ((const v16h*)base)[tile32 * 32 + (threadIdx.x & 31)];
}

// A tile for the 1x1 residual conv gathered from fp32 x in LDS:
// element (r, f) = x[n=r/5][f][t=r%5], rows >= 190 are zero.
__device__ inline v16h load_a_res(const float* Xf, int r0, int k0) {
    int lane = threadIdx.x & 31;
    int m = lane & 15;
    int hi = lane >> 4;
    int r = r0 + m;
    v16h a;
#pragma unroll
    for (int j = 0; j < 8; ++j) {
        int kb = k0 + (j >> 2) * 16 + hi * 8 + (j & 3) * 2;
#pragma unroll
        for (int h = 0; h < 2; ++h) {
            int f = kb + h;
            float v = (r < N_ * T_) ? Xf[(r / T_) * FT + f * T_ + (r % T_)] : 0.f;
            a[2 * j + h] = (half_t)v;
        }
    }
    return a;
}

__device__ inline float sigm(float x) { return 1.f / (1.f + __expf(-x)); }

__global__ __launch_bounds__(256)
void astgcn_block_kernel(const float* __restrict__ x, const float* __restrict__ cheb,
                         const float* __restrict__ U1, const float* __restrict__ U2,
                         const float* __restrict__ U3, const float* __restrict__ b_e,
                         const float* __restrict__ V_e, const float* __restrict__ W1,
                         const float* __restrict__ W2, const float* __restrict__ W3,
                         const float* __restrict__ b_s, const float* __restrict__ V_s,
                         const float* __restrict__ Theta, const float* __restrict__ W_time,
                         const float* __restrict__ b_time, const float* __restrict__ W_res,
                         const float* __restrict__ b_res, const float* __restrict__ gamma,
                         const float* __restrict__ beta, float* __restrict__ out) {
    extern __shared__ char smem[];
    float*  Pf  = (float*)(smem + OFF_PF);
    float*  Sat = (float*)(smem + OFF_SAT);
    float*  Sp  = (float*)(smem + OFF_SP);
    float*  Msc = (float*)(smem + OFF_MISC);
    half_t* Xh  = (half_t*)(smem + OFF_XH);
    half_t* Yb  = (half_t*)(smem + OFF_YB);
    half_t* Ag  = (half_t*)(smem + OFF_AG);
    half_t* Thb = (half_t*)(smem + OFF_TH);
    half_t* Wrb = (half_t*)(smem + OFF_WRB);
    half_t* Mh  = (half_t*)(smem + OFF_MH);

    float* lhs1 = Msc;         // 320
    float* lhsB = Msc + 320;   // 190
    float* rhsB = Msc + 512;   // 190
    float* prod = Msc + 704;   // 25
    float* Ebuf = Msc + 736;   // 25
    float* tat  = Msc + 768;   // 25
    float* w1b  = Msc + 800;   // 5
    float* slB  = Msc + 832;   // 190
    float* sr0  = Msc + 1024;  // 190
    float* srB  = Msc + 1216;  // 190

    const int tid  = threadIdx.x;
    const int lane = tid & 31;
    const int wave = tid >> 5;
    const int b    = blockIdx.x;
    const float* xb = x + (size_t)b * (N_ * FT);

    // ---- phase 1: stage x[b] (vectorized), zero im2col buffer ----
    {
        const float4* xb4 = (const float4*)xb;
        float4* Pf4 = (float4*)Pf;
        for (int i = tid; i < (N_ * FT) / 4; i += 256) Pf4[i] = xb4[i];
        v16h z = {};
        v16h* Ag16 = (v16h*)Ag;
        for (int i = tid; i < (RP * KF) / 16; i += 256) Ag16[i] = z;
        if (b + 1 < B_) __builtin_prefetch(xb + N_ * FT, 0, 1);
    }
    __syncthreads();

    // ---- phase 2: swizzled f16 operand builds + first attention reductions ----
    for (int i = tid; i < 64 * FT; i += 256) {
        int m = i / FT, c = i % FT;
        Xh[bswz(m, c, 20)] = (m < N_) ? (half_t)Pf[m * FT + c] : (half_t)0.f;
    }
    for (int i = tid; i < KF * C_; i += 256) {
        int j = i / C_, o = i % C_;
        Thb[bswz(j, o, 4)] = (half_t)Theta[i];
    }
    for (int i = tid; i < F_ * C_; i += 256) {
        int f = i / C_, c = i % C_;
        Wrb[bswz(f, c, 4)] = (half_t)W_res[c * F_ + f];
    }
    for (int i = tid; i < T_ * F_; i += 256) {  // lhs1[t,f] = sum_n x[n,f,t] U1[n]
        int t = i / F_, f = i % F_;
        float s = 0.f;
        for (int n = 0; n < N_; ++n) s += Pf[n * FT + f * T_ + t] * U1[n];
        lhs1[i] = s;
    }
    for (int i = tid; i < N_ * T_; i += 256) {  // rhs[n,t], sr0[n,t]
        int n = i / T_, t = i % T_;
        float s1 = 0.f, s2 = 0.f;
        for (int f = 0; f < F_; ++f) {
            float v = Pf[n * FT + f * T_ + t];
            s1 += U3[f] * v;
            s2 += W3[f] * v;
        }
        rhsB[i] = s1;
        sr0[i] = s2;
    }
    __syncthreads();

    // ---- temporal attention ----
    for (int i = tid; i < T_ * N_; i += 256) {  // lhs[t,n] = lhs1[t,:] @ U2[:,n]
        int t = i / N_, n = i % N_;
        float s = 0.f;
        for (int f = 0; f < F_; ++f) s += lhs1[t * F_ + f] * U2[f * N_ + n];
        lhsB[i] = s;
    }
    __syncthreads();
    for (int i = tid; i < T_ * T_; i += 256) {  // prod[t,s]
        int t = i / T_, s = i % T_;
        float acc = 0.f;
        for (int n = 0; n < N_; ++n) acc += lhsB[t * N_ + n] * rhsB[n * T_ + s];
        prod[i] = acc;
    }
    __syncthreads();
    for (int i = tid; i < T_ * T_; i += 256) {  // E[t,i2]
        int t = i / T_, i2 = i % T_;
        float acc = 0.f;
        for (int j = 0; j < T_; ++j)
            acc += sigm(prod[t * T_ + j] + b_e[t * T_ + j]) * V_e[i2 * T_ + j];
        Ebuf[i] = acc;
    }
    __syncthreads();
    if (tid < T_) {  // softmax over t (axis=1), column i2 = tid
        int i2 = tid;
        float mx = -1e30f;
        for (int t = 0; t < T_; ++t) mx = fmaxf(mx, Ebuf[t * T_ + i2]);
        float sm = 0.f;
        for (int t = 0; t < T_; ++t) sm += __expf(Ebuf[t * T_ + i2] - mx);
        for (int t = 0; t < T_; ++t) tat[t * T_ + i2] = __expf(Ebuf[t * T_ + i2] - mx) / sm;
    }
    __syncthreads();

    // ---- spatial attention (on x_tat, folded through t_at) ----
    if (tid < T_) {
        float s = 0.f;
        for (int j = 0; j < T_; ++j) s += tat[tid * T_ + j] * W1[j];
        w1b[tid] = s;
    }
    for (int i = tid; i < N_ * T_; i += 256) {  // sr[m,s] = sr0[m,:] @ t_at[:,s]
        int m = i / T_, s = i % T_;
        float acc = 0.f;
        for (int t = 0; t < T_; ++t) acc += sr0[m * T_ + t] * tat[t * T_ + s];
        srB[i] = acc;
    }
    __syncthreads();
    for (int i = tid; i < N_ * T_; i += 256) {  // sl[n,tt]
        int n = i / T_, tt = i % T_;
        float acc = 0.f;
        for (int f = 0; f < F_; ++f) {
            float s0 = 0.f;
            for (int t = 0; t < T_; ++t) s0 += Pf[n * FT + f * T_ + t] * w1b[t];
            acc += s0 * W2[f * T_ + tt];
        }
        slB[i] = acc;
    }
    __syncthreads();
    for (int i = tid; i < N_ * N_; i += 256) {  // sp[n,m]
        int n = i / N_, m = i % N_;
        float acc = 0.f;
        for (int t = 0; t < T_; ++t) acc += slB[n * T_ + t] * srB[m * T_ + t];
        Sp[i] = acc;
    }
    __syncthreads();
    for (int i = tid; i < N_ * N_; i += 256) {  // S[n,m] = V_s[n,:] @ sigm(sp+b_s)[:,m]
        int n = i / N_, m = i % N_;
        float acc = 0.f;
        for (int k = 0; k < N_; ++k)
            acc += V_s[n * N_ + k] * sigm(Sp[k * N_ + m] + b_s[k * N_ + m]);
        Sat[i] = acc;
    }
    __syncthreads();
    if (tid < N_) {  // softmax over n (axis=1), column m = tid, in place
        int m = tid;
        float mx = -1e30f;
        for (int n = 0; n < N_; ++n) mx = fmaxf(mx, Sat[n * N_ + m]);
        float sm = 0.f;
        for (int n = 0; n < N_; ++n) sm += __expf(Sat[n * N_ + m] - mx);
        float inv = 1.f / sm;
        for (int n = 0; n < N_; ++n) Sat[n * N_ + m] = __expf(Sat[n * N_ + m] - mx) * inv;
    }
    __syncthreads();

    // ---- phase E: Y_k = (cheb_k * s_at)^T @ X   via WMMA ----
    for (int k = 0; k < KCH; ++k) {
        for (int i = tid; i < NP * 64; i += 256) {
            int n = i >> 6, m = i & 63;
            float v = (n < N_ && m < N_) ? cheb[(k * N_ + m) * N_ + n] * Sat[m * N_ + n] : 0.f;
            Mh[aswz(n, m, 2)] = (half_t)v;
        }
        __syncthreads();
        for (int job = wave; job < 60; job += 8) {  // 3 n-tiles x 20 c-tiles
            int ti = job / 20, tj = job % 20;
            v8f acc = {};
#pragma unroll
            for (int kk = 0; kk < 2; ++kk) {
                v16h a  = load_frag(Mh, ti * 2 + kk);
                v16h bb = load_frag(Xh, kk * 20 + tj);
                acc = __builtin_amdgcn_wmma_f32_16x16x32_f16(false, a, false, bb,
                                                             (short)0, acc, false, false);
            }
            int col = tj * 16 + (lane & 15);
            int f = col / T_, t = col % T_;
            int rbase = ti * 16 + ((lane >> 4) * 8);
            half_t* Yt = Yb + t * (NP * KF);
#pragma unroll
            for (int j = 0; j < 8; ++j)
                Yt[aswz(rbase + j, k * F_ + f, 6)] = (half_t)acc[j];
        }
        __syncthreads();
    }

    // ---- phase F: gcn = relu(Y @ Theta), scattered into swizzled im2col A_g ----
    for (int job = wave; job < 60; job += 8) {  // 5 t x (3 n-tiles x 4 o-tiles)
        int s = job / 12, tile = job % 12;
        int ti = tile / 4, tj = tile % 4;
        v8f acc = {};
#pragma unroll
        for (int kk = 0; kk < 6; ++kk) {
            v16h a  = load_frag(Yb + s * (NP * KF), ti * 6 + kk);
            v16h bb = load_frag(Thb, kk * 4 + tj);
            acc = __builtin_amdgcn_wmma_f32_16x16x32_f16(false, a, false, bb,
                                                         (short)0, acc, false, false);
        }
        int o = tj * 16 + (lane & 15);
        int rbase = ti * 16 + ((lane >> 4) * 8);
#pragma unroll
        for (int j = 0; j < 8; ++j) {
            int n = rbase + j;
            if (n < N_) {
                half_t hv = (half_t)fmaxf(acc[j], 0.f);
#pragma unroll
                for (int dt = 0; dt < 3; ++dt) {
                    int tt = s + 1 - dt;
                    if (tt >= 0 && tt < T_)
                        Ag[aswz(n * T_ + tt, o * 3 + dt, 6)] = hv;
                }
            }
        }
    }
    __syncthreads();
    // Reuse Theta buffer for W_time: element (j=o*3+dt, c)
    for (int i = tid; i < KF * C_; i += 256) {
        int j = i / C_, c = i % C_;
        Thb[bswz(j, c, 4)] = (half_t)W_time[c * KF + j];
    }
    __syncthreads();

    // ---- phase G: temporal conv + residual 1x1, fused accumulate ----
    v8f accs[6];
#pragma unroll
    for (int q = 0; q < 6; ++q) {
        int job = wave * 6 + q;  // 12 row-tiles x 4 col-tiles = 48 jobs
        int i = job / 4, jx = job % 4;
        v8f acc = {};
#pragma unroll
        for (int kk = 0; kk < 6; ++kk) {
            v16h a  = load_frag(Ag, i * 6 + kk);
            v16h bb = load_frag(Thb, kk * 4 + jx);
            acc = __builtin_amdgcn_wmma_f32_16x16x32_f16(false, a, false, bb,
                                                         (short)0, acc, false, false);
        }
#pragma unroll
        for (int kk = 0; kk < 2; ++kk) {
            v16h a  = load_a_res(Pf, i * 16, kk * 32);
            v16h bb = load_frag(Wrb, kk * 4 + jx);
            acc = __builtin_amdgcn_wmma_f32_16x16x32_f16(false, a, false, bb,
                                                         (short)0, acc, false, false);
        }
        accs[q] = acc;
    }
    __syncthreads();  // everyone done reading x (Pf) before we overwrite it with h
#pragma unroll
    for (int q = 0; q < 6; ++q) {
        int job = wave * 6 + q;
        int r0 = (job / 4) * 16, c0 = (job % 4) * 16;
        int c = c0 + (lane & 15);
        float bias = b_time[c] + b_res[c];
        int rbase = r0 + ((lane >> 4) * 8);
#pragma unroll
        for (int j = 0; j < 8; ++j)
            Pf[(rbase + j) * C_ + c] = fmaxf(accs[q][j] + bias, 0.f);
    }
    __syncthreads();

    // ---- phase H: layernorm over channels, write (B,N,C,T) ----
    if (tid < N_ * T_) {
        int r = tid, n = r / T_, t = r % T_;
        float mu = 0.f;
        for (int c = 0; c < C_; ++c) mu += Pf[r * C_ + c];
        mu *= (1.f / C_);
        float var = 0.f;
        for (int c = 0; c < C_; ++c) {
            float d = Pf[r * C_ + c] - mu;
            var += d * d;
        }
        var *= (1.f / C_);
        float inv = rsqrtf(var + EPSL);
        float* ob = out + (((size_t)b * N_ + n) * C_) * T_ + t;
        for (int c = 0; c < C_; ++c)
            ob[c * T_] = gamma[c] * (Pf[r * C_ + c] - mu) * inv + beta[c];
    }
}

extern "C" void kernel_launch(void* const* d_in, const int* in_sizes, int n_in,
                              void* d_out, int out_size, void* d_ws, size_t ws_size,
                              hipStream_t stream) {
    (void)in_sizes; (void)n_in; (void)out_size; (void)d_ws; (void)ws_size;
    const float* x      = (const float*)d_in[0];
    const float* cheb   = (const float*)d_in[1];
    const float* U1     = (const float*)d_in[2];
    const float* U2     = (const float*)d_in[3];
    const float* U3     = (const float*)d_in[4];
    const float* b_e    = (const float*)d_in[5];
    const float* V_e    = (const float*)d_in[6];
    const float* W1     = (const float*)d_in[7];
    const float* W2     = (const float*)d_in[8];
    const float* W3     = (const float*)d_in[9];
    const float* b_s    = (const float*)d_in[10];
    const float* V_s    = (const float*)d_in[11];
    const float* Theta  = (const float*)d_in[12];
    const float* W_time = (const float*)d_in[13];
    const float* b_time = (const float*)d_in[14];
    const float* W_res  = (const float*)d_in[15];
    const float* b_res  = (const float*)d_in[16];
    const float* gamma  = (const float*)d_in[17];
    const float* beta   = (const float*)d_in[18];
    float* out = (float*)d_out;

    (void)hipFuncSetAttribute((const void*)astgcn_block_kernel,
                              hipFuncAttributeMaxDynamicSharedMemorySize, SMEM_SZ);
    astgcn_block_kernel<<<B_, 256, SMEM_SZ, stream>>>(
        x, cheb, U1, U2, U3, b_e, V_e, W1, W2, W3, b_s, V_s,
        Theta, W_time, b_time, W_res, b_res, gamma, beta, out);
}
